// DeepGraph_87866440942006
// MI455X (gfx1250) — matrix operand order repaired and checked by reading, not compiled, not run on hardware
//
#include <hip/hip_runtime.h>
#include <hip/hip_bf16.h>
#include <math.h>

// ---------------- problem constants (from reference) ----------------
#define NVERT   32768          // NN*BS
#define IN_C    32
#define HID     128
#define NLAYERS 4
#define NGROUP  64             // NN
#define BSZ     512            // BS
#define OUTC    3
#define NEDGE   524288
#define ETOT    (NEDGE + NVERT)   // edges + self loops = 557056
#define EPS_LN  1e-5f
#define EPS_SM  1e-16f

typedef __attribute__((ext_vector_type(16))) __bf16 v16bf;
typedef __attribute__((ext_vector_type(8)))  __bf16 v8bf;
typedef __attribute__((ext_vector_type(8)))  float  v8f;

// ---------------- helpers ----------------
__device__ __forceinline__ void atomicMaxFloat(float* addr, float val) {
    // monotonic int/uint mapping trick; addr initialized to -inf
    if (val >= 0.0f) atomicMax((int*)addr, __float_as_int(val));
    else             atomicMin((unsigned int*)addr, __float_as_uint(val));
}

__device__ __forceinline__ float lrelu02(float v) {
    return v > 0.0f ? v : 0.2f * v;
}

// =====================================================================
// LDS-staged WMMA GEMM for node features: C[M,128] = A[M,K] @ B[K,128].
// One block (8 waves) = one 16-row slab; wave w computes column tile w.
// A tile and B (transposed) are staged in LDS as bf16 with coalesced
// float4 global loads; fragments are contiguous ds_load_b128 reads.
// =====================================================================
template<int K>
__global__ void gemm_node_wmma(const float* __restrict__ A,
                               const float* __restrict__ B,   // [K,128] row-major
                               float* __restrict__ C)         // [M,128]
{
    __shared__ __bf16 As[16 * K];        // row-major, row stride K
    __shared__ __bf16 Bs[128 * K];       // transposed: Bs[n*K + k]

    const int tid  = threadIdx.x;
    const int lane = tid & 31;
    const int wave = tid >> 5;           // 0..7 -> column tile
    const int row0 = blockIdx.x << 4;

    // ---- stage A tile (16 x K f32 -> bf16), coalesced b128 loads ----
    {
        const int nvec = (16 * K) / 4;
        const float4* src = (const float4*)(A + (long)row0 * K);
#pragma unroll 2
        for (int v = tid; v < nvec; v += 256) {
            const float4 f = src[v];
            __bf16* dst = As + v * 4;
            dst[0] = (__bf16)f.x; dst[1] = (__bf16)f.y;
            dst[2] = (__bf16)f.z; dst[3] = (__bf16)f.w;
        }
    }
    // ---- stage B transposed (B[k][n] -> Bs[n*K+k]) ----
    {
        const int nvec = (K * 128) / 4;
        const float4* src = (const float4*)B;
        for (int v = tid; v < nvec; v += 256) {
            const float4 f = src[v];
            const int k = (v * 4) >> 7;       // / 128
            const int n = (v * 4) & 127;
            Bs[(n + 0) * K + k] = (__bf16)f.x;
            Bs[(n + 1) * K + k] = (__bf16)f.y;
            Bs[(n + 2) * K + k] = (__bf16)f.z;
            Bs[(n + 3) * K + k] = (__bf16)f.w;
        }
    }
    __syncthreads();

    // fragment addressing (ISA 7.12.2)
    const int mr  = lane & 15;           // A row within tile
    const int kha = (lane >> 4) * 16;    // A K-subgroup
    const int ncl = lane & 15;           // B/D column within tile
    const int nc  = (wave << 4) + ncl;   // global column
    const int khb = (lane >> 4) * 16;    // B K-half

    v8f acc = {};
#pragma unroll
    for (int k0 = 0; k0 < K; k0 += 32) {
        const __bf16* ap = As + mr * K + k0 + kha;      // 16B aligned
        const v8bf a0 = *(const v8bf*)(ap);
        const v8bf a1 = *(const v8bf*)(ap + 8);
        const v16bf afrag = __builtin_shufflevector(
            a0, a1, 0, 1, 2, 3, 4, 5, 6, 7, 8, 9, 10, 11, 12, 13, 14, 15);
        const v16bf bfrag = *(const v16bf*)(Bs + nc * K + k0 + khb); // 32B aligned
        acc = __builtin_amdgcn_wmma_f32_16x16x32_bf16(
            /*neg_a=*/false, afrag, /*neg_b=*/false, bfrag,
            /*c_mod=*/(short)0, acc, /*reuse_a=*/false, /*reuse_b=*/false);
    }

    // C/D layout: lane 0-15 -> M=row0+r ; lane 16-31 -> M=row0+8+r
    const int mb = row0 + (lane >> 4) * 8;
#pragma unroll
    for (int r = 0; r < 8; ++r)
        C[(long)(mb + r) * 128 + nc] = acc[r];
}

// =====================================================================
// Generic batched WMMA GEMM (used for predictor heads):
// C[b] = A[b] @ B[b];  A:[M,K], B:[K,Ncol], C:[M,Ncol] row-major.
// =====================================================================
template<int K>
__global__ void gemm_bf16_wmma(const float* __restrict__ A,
                               const float* __restrict__ B,
                               float* __restrict__ C,
                               int Ncol,
                               long strideA, long strideB, long strideC,
                               int tilesM, int tilesN, int relu)
{
    const int wave = (blockIdx.x * blockDim.x + threadIdx.x) >> 5;
    const int lane = threadIdx.x & 31;
    const int tilesPerBatch = tilesM * tilesN;
    const int b  = wave / tilesPerBatch;
    const int t  = wave % tilesPerBatch;
    const int tm = t / tilesN;
    const int tn = t % tilesN;

    const float* Ab = A + (long)b * strideA;
    const float* Bb = B + (long)b * strideB;
    float*       Cb = C + (long)b * strideC;

    const int row0 = tm << 4;
    const int col0 = tn << 4;

    const int mr  = row0 + (lane & 15);
    const int kha = (lane >> 4) * 16;
    const int nc  = col0 + (lane & 15);
    const int khb = (lane >> 4) * 16;

    v8f acc = {};
#pragma unroll
    for (int k0 = 0; k0 < K; k0 += 32) {
        v16bf afrag, bfrag;
        const float* arow = Ab + (long)mr * K + k0 + kha;
#pragma unroll
        for (int e = 0; e < 8; ++e) {
            afrag[e]     = (__bf16)arow[e];
            afrag[e + 8] = (__bf16)arow[8 + e];
        }
        const float* bcol = Bb + (long)(k0 + khb) * Ncol + nc;
#pragma unroll
        for (int e = 0; e < 16; ++e) {
            bfrag[e] = (__bf16)bcol[(long)e * Ncol];
        }
        acc = __builtin_amdgcn_wmma_f32_16x16x32_bf16(
            false, afrag, false, bfrag, (short)0, acc, false, false);
    }

    const int mb = row0 + (lane >> 4) * 8;
#pragma unroll
    for (int r = 0; r < 8; ++r) {
        float v = acc[r];
        if (relu) v = v > 0.0f ? v : 0.0f;
        Cb[(long)(mb + r) * Ncol + nc] = v;
    }
}

// ---------------- per-node attention scores (wave per node) ----------------
__global__ void score_kernel(const float* __restrict__ hw,
                             const float* __restrict__ asrc,
                             const float* __restrict__ adst,
                             float* __restrict__ a_s,
                             float* __restrict__ a_d)
{
    const int wid  = (blockIdx.x * blockDim.x + threadIdx.x) >> 5;
    const int lane = threadIdx.x & 31;
    if (wid >= NVERT) return;
    const float* row = hw + (long)wid * HID;
    float ps = 0.0f, pd = 0.0f;
#pragma unroll
    for (int j = 0; j < 4; ++j) {
        const int c = lane + j * 32;
        const float h = row[c];
        ps += h * asrc[c];
        pd += h * adst[c];
    }
#pragma unroll
    for (int off = 16; off > 0; off >>= 1) {
        ps += __shfl_xor(ps, off, 32);
        pd += __shfl_xor(pd, off, 32);
    }
    if (lane == 0) { a_s[wid] = ps; a_d[wid] = pd; }
}

// ---------------- init agg=0, m=-inf, z=0 ----------------
__global__ void init_kernel(float* __restrict__ agg,
                            float* __restrict__ m,
                            float* __restrict__ z)
{
    const int i = blockIdx.x * blockDim.x + threadIdx.x;
    if (i < NVERT * HID) agg[i] = 0.0f;
    if (i < NVERT) { m[i] = -INFINITY; z[i] = 0.0f; }
}

// ---------------- edge pass 1: segment max ----------------
__global__ void edge_max_kernel(const long long* __restrict__ ei,
                                const float* __restrict__ a_s,
                                const float* __restrict__ a_d,
                                float* __restrict__ m)
{
    const int e = blockIdx.x * blockDim.x + threadIdx.x;
    if (e >= ETOT) return;
    int s, d;
    if (e < NEDGE) { s = (int)ei[e]; d = (int)ei[NEDGE + e]; }
    else           { s = d = e - NEDGE; }
    atomicMaxFloat(&m[d], lrelu02(a_s[s] + a_d[d]));
}

// ---------------- edge pass 2: exp-sum ----------------
__global__ void edge_expsum_kernel(const long long* __restrict__ ei,
                                   const float* __restrict__ a_s,
                                   const float* __restrict__ a_d,
                                   const float* __restrict__ m,
                                   float* __restrict__ z)
{
    const int e = blockIdx.x * blockDim.x + threadIdx.x;
    if (e >= ETOT) return;
    int s, d;
    if (e < NEDGE) { s = (int)ei[e]; d = (int)ei[NEDGE + e]; }
    else           { s = d = e - NEDGE; }
    const float v = lrelu02(a_s[s] + a_d[d]);
    atomicAdd(&z[d], __expf(v - m[d]));
}

// ---------------- edge pass 3: weighted aggregation (wave per edge) -------
__global__ void edge_agg_kernel(const long long* __restrict__ ei,
                                const float* __restrict__ a_s,
                                const float* __restrict__ a_d,
                                const float* __restrict__ m,
                                const float* __restrict__ z,
                                const float* __restrict__ hw,
                                float* __restrict__ agg)
{
    const int wid  = (blockIdx.x * blockDim.x + threadIdx.x) >> 5;
    const int lane = threadIdx.x & 31;
    if (wid >= ETOT) return;
    int s, d;
    if (wid < NEDGE) { s = (int)ei[wid]; d = (int)ei[NEDGE + wid]; }
    else             { s = d = wid - NEDGE; }
    const float v     = lrelu02(a_s[s] + a_d[d]);
    const float alpha = __expf(v - m[d]) / (z[d] + EPS_SM);
    const float* hrow = hw  + (long)s * HID;
    float*       arow = agg + (long)d * HID;
#pragma unroll
    for (int j = 0; j < 4; ++j) {
        const int c = lane + j * 32;
        atomicAdd(&arow[c], hrow[c] * alpha);
    }
}

// ---------------- epilogues ----------------
__global__ void epi_first_kernel(const float* __restrict__ agg,
                                 const float* __restrict__ bias,
                                 float* __restrict__ h)
{
    const int i = blockIdx.x * blockDim.x + threadIdx.x;
    if (i >= NVERT * HID) return;
    const float v = agg[i] + bias[i & (HID - 1)];
    h[i] = v > 0.0f ? v : 0.0f;
}

__global__ void epi_res_kernel(const float* __restrict__ agg,
                               const float* __restrict__ bias,
                               float* __restrict__ h)
{
    const int i = blockIdx.x * blockDim.x + threadIdx.x;
    if (i >= NVERT * HID) return;
    h[i] += agg[i] + bias[i & (HID - 1)];
}

// ---------------- LayerNorm + ReLU (wave per row) ----------------
__global__ void ln_relu_kernel(const float* __restrict__ h,
                               const float* __restrict__ gamma,
                               const float* __restrict__ beta,
                               float* __restrict__ out)
{
    const int wid  = (blockIdx.x * blockDim.x + threadIdx.x) >> 5;
    const int lane = threadIdx.x & 31;
    if (wid >= NVERT) return;
    const float4* r4 = (const float4*)(h + (long)wid * HID);
    float4 v = r4[lane];
    float s = v.x + v.y + v.z + v.w;
#pragma unroll
    for (int off = 16; off > 0; off >>= 1) s += __shfl_xor(s, off, 32);
    const float mu = s * (1.0f / HID);
    float4 c;
    c.x = v.x - mu; c.y = v.y - mu; c.z = v.z - mu; c.w = v.w - mu;
    float q = c.x * c.x + c.y * c.y + c.z * c.z + c.w * c.w;
#pragma unroll
    for (int off = 16; off > 0; off >>= 1) q += __shfl_xor(q, off, 32);
    const float inv = rsqrtf(q * (1.0f / HID) + EPS_LN);
    const float4 g = ((const float4*)gamma)[lane];
    const float4 b = ((const float4*)beta)[lane];
    float4 o;
    o.x = c.x * inv * g.x + b.x;
    o.y = c.y * inv * g.y + b.y;
    o.z = c.z * inv * g.z + b.z;
    o.w = c.w * inv * g.w + b.w;
    o.x = o.x > 0.0f ? o.x : 0.0f;
    o.y = o.y > 0.0f ? o.y : 0.0f;
    o.z = o.z > 0.0f ? o.z : 0.0f;
    o.w = o.w > 0.0f ? o.w : 0.0f;
    ((float4*)(out + (long)wid * HID))[lane] = o;
}

// ---------------- predictor second matmul + softmax ----------------
__global__ void pred_logits_kernel(const float* __restrict__ t1,
                                   const float* __restrict__ W2p,
                                   float* __restrict__ out)
{
    const int i = blockIdx.x * blockDim.x + threadIdx.x;   // i in [0, NGROUP*BSZ)
    if (i >= NGROUP * BSZ) return;
    const int n = i >> 9;                                  // / BSZ
    const float* tr = t1  + (long)i * (HID / 2);
    const float* w  = W2p + (long)n * (HID / 2) * OUTC;
    float l0 = 0.0f, l1 = 0.0f, l2 = 0.0f;
#pragma unroll 8
    for (int k = 0; k < HID / 2; ++k) {
        const float tv = tr[k];
        l0 += tv * w[3 * k + 0];
        l1 += tv * w[3 * k + 1];
        l2 += tv * w[3 * k + 2];
    }
    const float mx = fmaxf(l0, fmaxf(l1, l2));
    const float e0 = __expf(l0 - mx), e1 = __expf(l1 - mx), e2 = __expf(l2 - mx);
    const float inv = 1.0f / (e0 + e1 + e2);
    out[3 * i + 0] = e0 * inv;
    out[3 * i + 1] = e1 * inv;
    out[3 * i + 2] = e2 * inv;
}

// =====================================================================
extern "C" void kernel_launch(void* const* d_in, const int* in_sizes, int n_in,
                              void* d_out, int out_size, void* d_ws, size_t ws_size,
                              hipStream_t stream) {
    (void)in_sizes; (void)n_in; (void)out_size; (void)ws_size;

    const float*     x    = (const float*)d_in[0];
    const long long* ei   = (const long long*)d_in[1];   // int64 edge_index [2,E]
    const float*     W0   = (const float*)d_in[2];
    const float*     as0  = (const float*)d_in[3];
    const float*     ad0  = (const float*)d_in[4];
    const float*     b0   = (const float*)d_in[5];
    const float*     Wres = (const float*)d_in[6];
    const float*     asr  = (const float*)d_in[7];
    const float*     adr  = (const float*)d_in[8];
    const float*     br   = (const float*)d_in[9];
    const float*     gam  = (const float*)d_in[10];
    const float*     bet  = (const float*)d_in[11];
    const float*     W1p  = (const float*)d_in[12];
    const float*     W2p  = (const float*)d_in[13];
    float*           out  = (float*)d_out;

    // workspace layout (floats)
    float* hw  = (float*)d_ws;                         // [N,128] gemm output
    float* agg = hw  + (long)NVERT * HID;              // [N,128] edge aggregation
    float* h   = agg + (long)NVERT * HID;              // [N,128] features
    float* t   = h   + (long)NVERT * HID;              // [N,128] ln/relu output
    float* t1  = t   + (long)NVERT * HID;              // [64,512,64] predictor hidden
    float* a_s = t1  + (long)NGROUP * BSZ * (HID / 2); // [N]
    float* a_d = a_s + NVERT;                          // [N]
    float* mbx = a_d + NVERT;                          // [N]
    float* zbx = mbx + NVERT;                          // [N]

    const dim3 blk(256);
    const int  nhidBlocks  = (NVERT * HID) / 256;      // 16384
    const int  waveNBlocks = NVERT / 8;                // 4096 (wave-per-node)
    const int  edgeBlocks  = ETOT / 256;               // 2176
    const int  edgeWBlocks = ETOT / 8;                 // 69632 (wave-per-edge)
    const int  slabBlocks  = NVERT / 16;               // 2048 (one 16-row slab/block)

    // ---- layer 0: GAT(x; W0) + relu ----
    {
        gemm_node_wmma<IN_C><<<slabBlocks, blk, 0, stream>>>(x, W0, hw);
        score_kernel<<<waveNBlocks, blk, 0, stream>>>(hw, as0, ad0, a_s, a_d);
        init_kernel<<<nhidBlocks, blk, 0, stream>>>(agg, mbx, zbx);
        edge_max_kernel<<<edgeBlocks, blk, 0, stream>>>(ei, a_s, a_d, mbx);
        edge_expsum_kernel<<<edgeBlocks, blk, 0, stream>>>(ei, a_s, a_d, mbx, zbx);
        edge_agg_kernel<<<edgeWBlocks, blk, 0, stream>>>(ei, a_s, a_d, mbx, zbx, hw, agg);
        epi_first_kernel<<<nhidBlocks, blk, 0, stream>>>(agg, b0, h);
    }

    // ---- residual blocks: h = h + GAT(relu(LN(h))) ----
    for (int i = 0; i < NLAYERS; ++i) {
        ln_relu_kernel<<<waveNBlocks, blk, 0, stream>>>(h, gam + i * HID, bet + i * HID, t);
        gemm_node_wmma<HID><<<slabBlocks, blk, 0, stream>>>(
            t, Wres + (long)i * HID * HID, hw);
        score_kernel<<<waveNBlocks, blk, 0, stream>>>(hw, asr + i * HID, adr + i * HID, a_s, a_d);
        init_kernel<<<nhidBlocks, blk, 0, stream>>>(agg, mbx, zbx);
        edge_max_kernel<<<edgeBlocks, blk, 0, stream>>>(ei, a_s, a_d, mbx);
        edge_expsum_kernel<<<edgeBlocks, blk, 0, stream>>>(ei, a_s, a_d, mbx, zbx);
        edge_agg_kernel<<<edgeWBlocks, blk, 0, stream>>>(ei, a_s, a_d, mbx, zbx, hw, agg);
        epi_res_kernel<<<nhidBlocks, blk, 0, stream>>>(agg, br + i * HID, h);
    }

    // ---- final LN (gamma[0], beta[0]) + relu ----
    ln_relu_kernel<<<waveNBlocks, blk, 0, stream>>>(h, gam, bet, t);

    // ---- predictor heads: t1[n] = relu(t[n] @ W1p[n]) via batched WMMA ----
    {
        const int tilesM = BSZ / 16, tilesN = (HID / 2) / 16;       // 32 x 4, batch 64
        gemm_bf16_wmma<HID><<<(NGROUP * tilesM * tilesN) / 8, blk, 0, stream>>>(
            t, W1p, t1, HID / 2,
            (long)BSZ * HID, (long)HID * (HID / 2), (long)BSZ * (HID / 2),
            tilesM, tilesN, 1);
    }

    // ---- logits + softmax ----
    pred_logits_kernel<<<(NGROUP * BSZ) / 256, blk, 0, stream>>>(t1, W2p, out);
}
